// VectorQuantizer_33758442946745
// MI455X (gfx1250) — matrix operand order repaired and checked by reading, not compiled, not run on hardware
//
#include <hip/hip_runtime.h>
#include <hip/hip_bf16.h>
#include <float.h>

typedef __attribute__((ext_vector_type(16))) _Float16 v16h;
typedef __attribute__((ext_vector_type(2)))  _Float16 v2h;
typedef __attribute__((ext_vector_type(8)))  float    v8f;
typedef __attribute__((ext_vector_type(2)))  float    v2f;

#define NUM_EMB         81920
#define EMB_DIM         64
#define NROWS           2048
#define NCHUNK          8
#define TILES_PER_CHUNK 640     /* 81920 codes / 16 per tile / 8 chunks */
#define CODES_PER_CHUNK 10240
#define Q_ELEMS         131072  /* 2048*64 */
/* d_out layout: [0,131072) quantized_st | 131072 vq_loss | 131073 commit_loss | [131074,133122) idx */

/* workspace layout (bytes) */
#define EMBH_OFF 0            /* f16 codebook * 2^16 : 81920*64*2 = 10485760 */
#define NORM_OFF 10485760     /* 2^15*||e||^2 fp32   : 81920*4    = 327680   */
#define PVAL_OFF 10813440     /* per-(row,chunk) max : 2048*8*4              */
#define PIDX_OFF 10878976
#define BEST_OFF 10944512
#define BSUM_OFF 10952704

// ---------------------------------------------------------------------------
// Kernel 0: one-time codebook prep. One wave per code row:
//   emb_h[k][d] = f16(emb[k][d] * 2^16)   (power-of-two scale keeps the
//   uniform(+-1/81920) values out of f16-subnormal range, argmin invariant)
//   norms[k]    = 2^15 * ||e_k||^2  in fp32
// ---------------------------------------------------------------------------
__global__ __launch_bounds__(256) void vq_prep_kernel(
    const float* __restrict__ emb, _Float16* __restrict__ emb_h,
    float* __restrict__ norms)
{
  const int wid  = (blockIdx.x * 256 + threadIdx.x) >> 5;   // code row
  const int lane = threadIdx.x & 31;
  const float* er = emb + (size_t)wid * EMB_DIM + lane * 2;
  v2f q = *(const v2f*)er;
  v2h h;
  h[0] = (_Float16)(q[0] * 65536.0f);
  h[1] = (_Float16)(q[1] * 65536.0f);
  *(v2h*)(emb_h + (size_t)wid * EMB_DIM + lane * 2) = h;
  float ssq = q[0] * q[0] + q[1] * q[1];
  #pragma unroll
  for (int mask = 1; mask <= 16; mask <<= 1) ssq += __shfl_xor(ssq, mask);
  if (lane == 0) norms[wid] = ssq * 32768.0f;
}

// ---------------------------------------------------------------------------
// Kernel 1: fused scaled-GEMM + running argmax per (16-row tile, code chunk).
// score'_k = 2^16*(z.e_k) - 2^15*||e_k||^2 ; argmax score' == argmin dist.
// FOUR independent 16-code tiles per iteration: each WMMA's 4-slot RAW
// hazard window (f16 WMMA -> VALU, ISA 7.12.1) is filled by the other
// chains' WMMAs / compare-select work instead of v_nops.
// ---------------------------------------------------------------------------
__global__ __launch_bounds__(256) void vq_argmax_kernel(
    const float* __restrict__ z, const _Float16* __restrict__ emb_h,
    const float* __restrict__ norms,
    float* __restrict__ pval, int* __restrict__ pidx)
{
  const int mtile = blockIdx.x;          // 0..127 (16 z-rows each)
  const int chunk = blockIdx.y;          // 0..7   (10240 codes each)
  const int wave  = threadIdx.x >> 5;    // 0..7
  const int lane  = threadIdx.x & 31;
  const int col   = lane & 15;           // WMMA column / row-in-tile
  const int hi    = lane >> 4;           // lane half selects K-subrange

  // ---- A operand: z tile (16x64), f16, contiguous K order (same as B) ----
  // lanes 0-15: K = 0..15; lanes 16-31: K = 16..31  (per 16x32 block)
  const int row = mtile * 16 + col;
  const int d0  = hi ? 16 : 0;
  const float* zr = z + (size_t)row * EMB_DIM;
  v16h a0, a1;
  #pragma unroll
  for (int j = 0; j < 8; ++j) {
    a0[j]     = (_Float16)zr[d0 + j];
    a0[8 + j] = (_Float16)zr[d0 + 8 + j];
    a1[j]     = (_Float16)zr[d0 + 32 + j];
    a1[8 + j] = (_Float16)zr[d0 + 40 + j];
  }

  float vmax[8];
  int   imax[8];
  #pragma unroll
  for (int r = 0; r < 8; ++r) { vmax[r] = -FLT_MAX; imax[r] = 0; }

  // ---- B operand: lanes 0-15 hold K=0..15, lanes 16-31 hold K=16..31 ----
  const int e0 = hi ? 16 : 0;
  const int t0 = wave * 4;                      // first tile of this wave
  const _Float16* eb = emb_h +
      (size_t)(chunk * CODES_PER_CHUNK + t0 * 16 + col) * EMB_DIM + e0;
  const float* nb = norms + chunk * CODES_PER_CHUNK + t0 * 16 + col;
  int code0 = chunk * CODES_PER_CHUNK + t0 * 16 + col;

  for (int i = 0; i < TILES_PER_CHUNK / 32; ++i) {         // 20 iterations
    v16h b0[4], b1[4];
    float n[4];
    #pragma unroll
    for (int j = 0; j < 4; ++j) {
      b0[j] = *(const v16h*)(eb + j * 16 * EMB_DIM);        // tile j, K 0..31
      b1[j] = *(const v16h*)(eb + j * 16 * EMB_DIM + 32);   // tile j, K 32..63
      n[j]  = nb[j * 16];
    }

    if (i + 1 < TILES_PER_CHUNK / 32)
      __builtin_prefetch((const void*)(eb + 512 * EMB_DIM), 0, 0);

    v8f acc[4];
    #pragma unroll
    for (int j = 0; j < 4; ++j) {
      #pragma unroll
      for (int r = 0; r < 8; ++r) acc[j][r] = -n[j];
    }

    #pragma unroll
    for (int j = 0; j < 4; ++j)
      acc[j] = __builtin_amdgcn_wmma_f32_16x16x32_f16(false, a0, false, b0[j],
                                                      (short)0, acc[j], false, false);
    #pragma unroll
    for (int j = 0; j < 4; ++j)
      acc[j] = __builtin_amdgcn_wmma_f32_16x16x32_f16(false, a1, false, b1[j],
                                                      (short)0, acc[j], false, false);

    #pragma unroll
    for (int j = 0; j < 4; ++j) {          // ascending code order -> argmin tie
      const int cj = code0 + j * 16;
      #pragma unroll
      for (int r = 0; r < 8; ++r) {
        if (acc[j][r] > vmax[r]) { vmax[r] = acc[j][r]; imax[r] = cj; }
      }
    }

    eb    += 512 * EMB_DIM;   // 32 tiles * 16 codes ahead
    nb    += 512;
    code0 += 512;
  }

  // reduce the 16 columns within each lane-half (rows live per-VGPR)
  #pragma unroll
  for (int mask = 1; mask <= 8; mask <<= 1) {
    #pragma unroll
    for (int r = 0; r < 8; ++r) {
      float ov = __shfl_xor(vmax[r], mask);
      int   oi = __shfl_xor(imax[r], mask);
      if (ov > vmax[r] || (ov == vmax[r] && oi < imax[r])) {
        vmax[r] = ov; imax[r] = oi;
      }
    }
  }

  // combine the 8 waves of this block via LDS
  __shared__ float sval[8][16];
  __shared__ int   sidx[8][16];
  if (col == 0) {   // lane 0 holds rows 0..7, lane 16 holds rows 8..15
    #pragma unroll
    for (int r = 0; r < 8; ++r) {
      sval[wave][hi * 8 + r] = vmax[r];
      sidx[wave][hi * 8 + r] = imax[r];
    }
  }
  __syncthreads();
  if (threadIdx.x < 16) {
    float bv = -FLT_MAX; int bi = 0;
    #pragma unroll
    for (int w = 0; w < 8; ++w) {
      float v = sval[w][threadIdx.x];
      int   i = sidx[w][threadIdx.x];
      if (v > bv || (v == bv && i < bi)) { bv = v; bi = i; }
    }
    const int orow = mtile * 16 + threadIdx.x;
    pval[orow * NCHUNK + chunk] = bv;
    pidx[orow * NCHUNK + chunk] = bi;
  }
}

// ---------------------------------------------------------------------------
// Kernel 2: reduce 8 chunk partials per row -> final argmin index.
// ---------------------------------------------------------------------------
__global__ __launch_bounds__(256) void vq_pick_kernel(
    const float* __restrict__ pval, const int* __restrict__ pidx,
    int* __restrict__ best, float* __restrict__ out)
{
  const int m = blockIdx.x * blockDim.x + threadIdx.x;
  if (m >= NROWS) return;
  float bv = -FLT_MAX; int bi = 0;
  #pragma unroll
  for (int c = 0; c < NCHUNK; ++c) {
    float v = pval[m * NCHUNK + c];
    int   i = pidx[m * NCHUNK + c];
    if (v > bv || (v == bv && i < bi)) { bv = v; bi = i; }
  }
  best[m] = bi;
  out[Q_ELEMS + 2 + m] = (float)bi;   // idx output (as float)
}

// ---------------------------------------------------------------------------
// Kernel 3: gather quantized = emb[idx]; per-block deterministic loss sum.
// ---------------------------------------------------------------------------
__global__ __launch_bounds__(256) void vq_gather_loss_kernel(
    const float* __restrict__ z, const float* __restrict__ emb,
    const int* __restrict__ best, float* __restrict__ out,
    float* __restrict__ bsum)
{
  const int gid = blockIdx.x * 256 + threadIdx.x;   // 512 blocks cover 131072
  const int m = gid >> 6;
  const int d = gid & 63;
  const int k = best[m];
  const float q = emb[(size_t)k * EMB_DIM + d];
  out[gid] = q;                        // quantized_st forward value == quantized
  const float diff = z[gid] - q;
  __shared__ float red[256];
  red[threadIdx.x] = diff * diff;
  __syncthreads();
  #pragma unroll
  for (int off = 128; off > 0; off >>= 1) {
    if (threadIdx.x < off) red[threadIdx.x] += red[threadIdx.x + off];
    __syncthreads();
  }
  if (threadIdx.x == 0) bsum[blockIdx.x] = red[0];
}

// ---------------------------------------------------------------------------
// Kernel 4: fixed-order reduction of 512 block sums -> both loss scalars.
// ---------------------------------------------------------------------------
__global__ __launch_bounds__(256) void vq_final_loss_kernel(
    const float* __restrict__ bsum, float* __restrict__ out)
{
  __shared__ float red[256];
  red[threadIdx.x] = bsum[threadIdx.x] + bsum[threadIdx.x + 256];
  __syncthreads();
  #pragma unroll
  for (int off = 128; off > 0; off >>= 1) {
    if (threadIdx.x < off) red[threadIdx.x] += red[threadIdx.x + off];
    __syncthreads();
  }
  if (threadIdx.x == 0) {
    const float mean = red[0] / (float)Q_ELEMS;
    out[Q_ELEMS]     = mean;   // vq_loss (forward value)
    out[Q_ELEMS + 1] = mean;   // commitment_loss (identical forward value)
  }
}

// ---------------------------------------------------------------------------
extern "C" void kernel_launch(void* const* d_in, const int* in_sizes, int n_in,
                              void* d_out, int out_size, void* d_ws, size_t ws_size,
                              hipStream_t stream)
{
  (void)in_sizes; (void)n_in; (void)out_size; (void)ws_size;
  const float* z   = (const float*)d_in[0];   // [2048, 64]
  const float* emb = (const float*)d_in[1];   // [81920, 64]
  float* out = (float*)d_out;

  char* ws = (char*)d_ws;
  _Float16* emb_h = (_Float16*)(ws + EMBH_OFF);
  float*    norms = (float*)   (ws + NORM_OFF);
  float*    pval  = (float*)   (ws + PVAL_OFF);
  int*      pidx  = (int*)     (ws + PIDX_OFF);
  int*      best  = (int*)     (ws + BEST_OFF);
  float*    bsum  = (float*)   (ws + BSUM_OFF);

  vq_prep_kernel<<<NUM_EMB / 8, 256, 0, stream>>>(emb, emb_h, norms);
  vq_argmax_kernel<<<dim3(128, NCHUNK), 256, 0, stream>>>(z, emb_h, norms, pval, pidx);
  vq_pick_kernel<<<NROWS / 256, 256, 0, stream>>>(pval, pidx, best, out);
  vq_gather_loss_kernel<<<Q_ELEMS / 256, 256, 0, stream>>>(z, emb, best, out, bsum);
  vq_final_loss_kernel<<<1, 256, 0, stream>>>(bsum, out);
}
